// PerPartEncoderTpl_85066122265630
// MI455X (gfx1250) — compile-verified, matrix operand order checked by reading, not run on hardware
//
#include <hip/hip_runtime.h>
#include <hip/hip_bf16.h>

typedef __attribute__((ext_vector_type(16))) _Float16 v16h;
typedef __attribute__((ext_vector_type(8)))  _Float16 v8h;
typedef __attribute__((ext_vector_type(8)))  float    v8f;

#define EPSF 1e-5f

#define SHUF16(lo, hi) __builtin_shufflevector(lo, hi, 0, 1, 2, 3, 4, 5, 6, 7, \
                                               8, 9, 10, 11, 12, 13, 14, 15)

// ---------------------------------------------------------------------------
// fragment helpers (16x64 tile per wave: 1 A fragment + 4 B fragments)
// ---------------------------------------------------------------------------
struct Frag {
  v8h a0, a1;
  v8h b0[4], b1[4];
};

__device__ __forceinline__ void load_frag(Frag& f, const _Float16* arow,
                                          const _Float16* const* wrow, int k) {
  f.a0 = *(const v8h*)(arow + k);
  f.a1 = *(const v8h*)(arow + k + 16);
#pragma unroll
  for (int j = 0; j < 4; ++j) {
    f.b0[j] = *(const v8h*)(wrow[j] + k);
    f.b1[j] = *(const v8h*)(wrow[j] + k + 16);
  }
}

__device__ __forceinline__ void mma_frag(v8f* acc, const Frag& f) {
  v16h a = SHUF16(f.a0, f.a1);
#pragma unroll
  for (int j = 0; j < 4; ++j) {
    v16h b = SHUF16(f.b0[j], f.b1[j]);
    acc[j] = __builtin_amdgcn_wmma_f32_16x16x32_f16(false, a, false, b, (short)0, acc[j],
                                                    false, false);
  }
}

// EdgeConv variant: A rows are the virtual matrix [x_i | x_j - x_i]
__device__ __forceinline__ void load_efrag(Frag& f, const _Float16* Xd, const _Float16* Xs,
                                           int Cin, int halfOfs,
                                           const _Float16* const* wrow, int k) {
#pragma unroll
  for (int p = 0; p < 2; ++p) {
    int kk = k + p * 16 + halfOfs;
    bool second = kk >= Cin;
    int base = second ? (kk - Cin) : kk;
    v8h vd = *(const v8h*)(Xd + base);
    v8h vs = *(const v8h*)(Xs + base);
    v8h val = second ? (v8h)(vs - vd) : vd;
    if (p == 0) f.a0 = val; else f.a1 = val;
  }
#pragma unroll
  for (int j = 0; j < 4; ++j) {
    f.b0[j] = *(const v8h*)(wrow[j] + k);
    f.b1[j] = *(const v8h*)(wrow[j] + k + 16);
  }
}

// ---------------------------------------------------------------------------
// utility kernels
// ---------------------------------------------------------------------------
__global__ void zero32_kernel(unsigned* p, long n) {
  long i = (long)blockIdx.x * blockDim.x + threadIdx.x;
  if (i < n) p[i] = 0u;
}

// f32 weights (rows x Kin) -> f16 (rows x KinPad), zero-padded
__global__ void cvt_w_kernel(const float* __restrict__ W, _Float16* __restrict__ Wh,
                             long n, int Kin, int KinPad) {
  long i = (long)blockIdx.x * blockDim.x + threadIdx.x;
  if (i >= n) return;
  long r = i / KinPad;
  int k = (int)(i % KinPad);
  Wh[i] = (k < Kin) ? (_Float16)W[r * Kin + k] : (_Float16)0.f;
}

// gcu1 edge features, zero-padded to 64: [xi(3) | xj-xi(3) | 0...]
__global__ void build_feat0_kernel(const float* __restrict__ pos, const int* __restrict__ srcI,
                                   const int* __restrict__ dstI, _Float16* __restrict__ feat,
                                   long E) {
  long i = (long)blockIdx.x * blockDim.x + threadIdx.x;
  if (i >= E * 64) return;
  long e = i >> 6;
  int c = (int)(i & 63);
  int di = dstI[e], si = srcI[e];
  int cc = (c < 3) ? c : ((c < 6) ? c - 3 : 0);
  float xi = pos[(long)di * 3 + cc];
  float xj = pos[(long)si * 3 + cc];
  float v = (c < 3) ? xi : ((c < 6) ? (xj - xi) : 0.f);
  feat[i] = (_Float16)v;
}

// stats = [sum(C), sumsq(C)] -> ss = [scale(C), shift(C)]
__global__ void bn_finalize_kernel(const float* __restrict__ stats,
                                   const float* __restrict__ gamma,
                                   const float* __restrict__ beta,
                                   float* __restrict__ ss, float invM, int C) {
  int c = blockIdx.x * blockDim.x + threadIdx.x;
  if (c >= C) return;
  float mu  = stats[c] * invM;
  float var = stats[C + c] * invM - mu * mu;
  float s   = gamma[c] * rsqrtf(var + EPSF);
  ss[c]     = s;
  ss[C + c] = beta[c] - mu * s;
}

// vectorized BN apply in place; n8 = total_elems/8, C multiple of 8
__global__ void bn_apply_kernel(_Float16* z, const float* __restrict__ ss, long n8, int C) {
  long i = (long)blockIdx.x * blockDim.x + threadIdx.x;
  if (i >= n8) return;
  long base = i * 8;
  int c = (int)(base % C);
  v8h v = *(const v8h*)(z + base);
#pragma unroll
  for (int j = 0; j < 8; ++j)
    v[j] = (_Float16)((float)v[j] * ss[c + j] + ss[C + c + j]);
  *(v8h*)(z + base) = v;
}

// BN-apply fused with segment-max scatter (ordered-uint encoding; init 0 == -inf)
__global__ void bn_scatter_max_kernel(const _Float16* __restrict__ z,
                                      const float* __restrict__ ss,
                                      const int* __restrict__ dstI,
                                      unsigned* __restrict__ enc, long Ecnt, int C) {
  long i = (long)blockIdx.x * blockDim.x + threadIdx.x;
  if (i >= Ecnt * C) return;
  long e = i / C;
  int  c = (int)(i % C);
  float v = (float)z[i] * ss[c] + ss[C + c];
  unsigned bits = __float_as_uint(v);
  unsigned key = (bits & 0x80000000u) ? ~bits : (bits | 0x80000000u);
  atomicMax(enc + (long)dstI[e] * C + c, key);
}

__global__ void decode_max_kernel(const unsigned* __restrict__ enc,
                                  _Float16* __restrict__ x, long n) {
  long i = (long)blockIdx.x * blockDim.x + threadIdx.x;
  if (i >= n) return;
  unsigned key = enc[i];
  float v = 0.f;
  if (key != 0u) {
    unsigned bits = (key & 0x80000000u) ? (key & 0x7FFFFFFFu) : ~key;
    if ((bits & 0x7F800000u) != 0x7F800000u) v = __uint_as_float(bits);  // finite only
  }
  x[i] = (_Float16)v;
}

// x123 = [x1(64)|x2(128)|x3(256)], vectorized by 8 (segment bounds are mult. of 8)
__global__ void concat3_kernel(const _Float16* __restrict__ a, const _Float16* __restrict__ b,
                               const _Float16* __restrict__ c, _Float16* __restrict__ out,
                               long n) {
  long i = (long)blockIdx.x * blockDim.x + threadIdx.x;  // chunk id, n*56 chunks
  if (i >= n * 56) return;
  long r = i / 56;
  int c8 = (int)(i % 56) * 8;
  v8h v;
  if (c8 < 64)       v = *(const v8h*)(a + r * 64 + c8);
  else if (c8 < 192) v = *(const v8h*)(b + r * 128 + (c8 - 64));
  else               v = *(const v8h*)(c + r * 256 + (c8 - 192));
  *(v8h*)(out + r * 448 + c8) = v;
}

// per-mesh max over V vertices of one channel; grid=(C, B), block=256
__global__ void mesh_maxpool_kernel(const _Float16* __restrict__ x, _Float16* __restrict__ out,
                                    int V, int C) {
  __shared__ float red[256];
  int c = blockIdx.x, b = blockIdx.y;
  float m = -3.0e38f;
  for (int v = threadIdx.x; v < V; v += blockDim.x)
    m = fmaxf(m, (float)x[((long)b * V + v) * C + c]);
  red[threadIdx.x] = m;
  __syncthreads();
  for (int s = 128; s > 0; s >>= 1) {
    if ((int)threadIdx.x < s) red[threadIdx.x] = fmaxf(red[threadIdx.x], red[threadIdx.x + s]);
    __syncthreads();
  }
  if (threadIdx.x == 0) out[(long)b * C + c] = (_Float16)red[0];
}

// x5 = [x4(256) | xg[batch](256)], vectorized by 8
__global__ void build_x5_kernel(const _Float16* __restrict__ x4, const _Float16* __restrict__ xg,
                                const int* __restrict__ batch, _Float16* __restrict__ x5,
                                long n) {
  long i = (long)blockIdx.x * blockDim.x + threadIdx.x;  // chunk id, n*64 chunks
  if (i >= n * 64) return;
  long r = i / 64;
  int c8 = (int)(i % 64) * 8;
  v8h v;
  if (c8 < 256) v = *(const v8h*)(x4 + r * 256 + c8);
  else          v = *(const v8h*)(xg + (long)batch[r] * 256 + (c8 - 256));
  *(v8h*)(x5 + r * 512 + c8) = v;
}

__global__ void final_linear_kernel(const float* __restrict__ X, const float* __restrict__ W,
                                    const float* __restrict__ bias, float* __restrict__ Z,
                                    float* __restrict__ stats, int M, int Kin, int Cout) {
  int i = blockIdx.x * blockDim.x + threadIdx.x;
  if (i >= M * Cout) return;
  int m = i / Cout, o = i % Cout;
  float acc = bias[o];
  const float* xr = X + (long)m * Kin;
  const float* wr = W + (long)o * Kin;
  for (int k = 0; k < Kin; ++k) acc += xr[k] * wr[k];
  acc = fmaxf(acc, 0.f);
  Z[i] = acc;
  atomicAdd(&stats[o], acc);
  atomicAdd(&stats[Cout + o], acc * acc);
}

__global__ void bn_out_kernel(const float* __restrict__ z, const float* __restrict__ ss,
                              float* __restrict__ out, int n, int C) {
  int i = blockIdx.x * blockDim.x + threadIdx.x;
  if (i >= n) return;
  int c = i % C;
  out[i] = z[i] * ss[c] + ss[C + c];
}

// ---------------------------------------------------------------------------
// WMMA GEMM: Z = relu(A @ Wh^T + bias). A f16 MxKin, Wh f16 CoutxKin.
// Wave computes a 16x64 tile. Ping-pong fragment sets (unroll-by-2, Kin%64==0)
// remove the double-buffer register copies: each set is loaded one half-
// iteration before its 4 WMMAs, so the compiler can emit partial waits.
// OOB rows clamped; results discarded at the store.
// block = 256 (8 waves) -> 128 x 64 per block; grid = (ceil(M/128), Cout/64)
// ---------------------------------------------------------------------------
__global__ __launch_bounds__(256) void gemm_bias_relu_stats(
    const _Float16* __restrict__ A, const _Float16* __restrict__ Wh,
    const float* __restrict__ bias, _Float16* __restrict__ Z,
    float* __restrict__ stats, long M, int Kin, int Cout) {
  __shared__ float s_sum[64];
  __shared__ float s_sq[64];
  if (threadIdx.x < 64) { s_sum[threadIdx.x] = 0.f; s_sq[threadIdx.x] = 0.f; }
  __syncthreads();
  int lane = threadIdx.x & 31, wave = threadIdx.x >> 5;
  long tileM = (long)blockIdx.x * 128 + wave * 16;
  int tileN = blockIdx.y * 64;
  int l16 = lane & 15, half = lane >> 4;
  long row = tileM + l16;
  long rc = (row < M) ? row : (M - 1);  // clamp: garbage rows never stored
  const _Float16* arow = A + rc * Kin + half * 8;
  const _Float16* wrow[4];
#pragma unroll
  for (int j = 0; j < 4; ++j)
    wrow[j] = Wh + (long)(tileN + j * 16 + l16) * Kin + half * 8;

  v8f acc[4] = {};
  Frag f0, f1;
  load_frag(f0, arow, wrow, 0);
  for (int k0 = 0; k0 < Kin; k0 += 64) {
    int k1 = (k0 + 32 < Kin) ? (k0 + 32) : 0;  // clamped: always in-bounds
    int k2 = (k0 + 64 < Kin) ? (k0 + 64) : 0;
    __builtin_prefetch(arow + ((k0 + 96 < Kin) ? (k0 + 96) : 0), 0, 1);
    load_frag(f1, arow, wrow, k1);
    mma_frag(acc, f0);
    load_frag(f0, arow, wrow, k2);
    mma_frag(acc, f1);
  }
#pragma unroll
  for (int j = 0; j < 4; ++j) {
    int col = tileN + j * 16 + l16;
    float bb = bias[col];
    float ls = 0.f, lq = 0.f;
#pragma unroll
    for (int r = 0; r < 8; ++r) {
      long mr = tileM + r + half * 8;  // D: VGPR r -> M=r (lanes 0-15), M=8+r (16-31)
      float v = acc[j][r] + bb;
      v = v > 0.f ? v : 0.f;
      if (mr < M) {
        Z[mr * Cout + col] = (_Float16)v;
        ls += v; lq += v * v;
      }
    }
    atomicAdd(&s_sum[j * 16 + l16], ls);
    atomicAdd(&s_sq[j * 16 + l16], lq);
  }
  __syncthreads();
  if (threadIdx.x < 64) {
    atomicAdd(&stats[tileN + threadIdx.x], s_sum[threadIdx.x]);
    atomicAdd(&stats[Cout + tileN + threadIdx.x], s_sq[threadIdx.x]);
  }
}

// ---------------------------------------------------------------------------
// EdgeConv first layer: A-operand is the virtual matrix [x_i | x_j - x_i]
// gathered on the fly. X f16 NxCin, Cin % 16 == 0 so 8-chunks never straddle
// the Cin boundary. Same ping-pong 16x64 structure; Kin = 2*Cin % 64 == 0.
// ---------------------------------------------------------------------------
__global__ __launch_bounds__(256) void edge_gemm_bias_relu_stats(
    const _Float16* __restrict__ X, const int* __restrict__ srcI,
    const int* __restrict__ dstI, const _Float16* __restrict__ Wh,
    const float* __restrict__ bias, _Float16* __restrict__ Z,
    float* __restrict__ stats, long M, int Cin, int Cout) {
  __shared__ float s_sum[64];
  __shared__ float s_sq[64];
  if (threadIdx.x < 64) { s_sum[threadIdx.x] = 0.f; s_sq[threadIdx.x] = 0.f; }
  __syncthreads();
  const int Kin = 2 * Cin;
  int lane = threadIdx.x & 31, wave = threadIdx.x >> 5;
  long tileM = (long)blockIdx.x * 128 + wave * 16;
  int tileN = blockIdx.y * 64;
  int l16 = lane & 15, half = lane >> 4;
  long row = tileM + l16;
  long rc = (row < M) ? row : (M - 1);
  int di = dstI[rc], si = srcI[rc];
  const _Float16* Xd = X + (long)di * Cin;
  const _Float16* Xs = X + (long)si * Cin;
  int halfOfs = half * 8;
  const _Float16* wrow[4];
#pragma unroll
  for (int j = 0; j < 4; ++j)
    wrow[j] = Wh + (long)(tileN + j * 16 + l16) * Kin + halfOfs;

  v8f acc[4] = {};
  Frag f0, f1;
  load_efrag(f0, Xd, Xs, Cin, halfOfs, wrow, 0);
  for (int k0 = 0; k0 < Kin; k0 += 64) {
    int k1 = (k0 + 32 < Kin) ? (k0 + 32) : 0;
    int k2 = (k0 + 64 < Kin) ? (k0 + 64) : 0;
    load_efrag(f1, Xd, Xs, Cin, halfOfs, wrow, k1);
    mma_frag(acc, f0);
    load_efrag(f0, Xd, Xs, Cin, halfOfs, wrow, k2);
    mma_frag(acc, f1);
  }
#pragma unroll
  for (int j = 0; j < 4; ++j) {
    int col = tileN + j * 16 + l16;
    float bb = bias[col];
    float ls = 0.f, lq = 0.f;
#pragma unroll
    for (int r = 0; r < 8; ++r) {
      long mr = tileM + r + half * 8;
      float v = acc[j][r] + bb;
      v = v > 0.f ? v : 0.f;
      if (mr < M) {
        Z[mr * Cout + col] = (_Float16)v;
        ls += v; lq += v * v;
      }
    }
    atomicAdd(&s_sum[j * 16 + l16], ls);
    atomicAdd(&s_sq[j * 16 + l16], lq);
  }
  __syncthreads();
  if (threadIdx.x < 64) {
    atomicAdd(&stats[tileN + threadIdx.x], s_sum[threadIdx.x]);
    atomicAdd(&stats[Cout + tileN + threadIdx.x], s_sq[threadIdx.x]);
  }
}

// ---------------------------------------------------------------------------
// x6[b,k,c] = sum_v hm[b,v,k] * x5[b,v,c]  (per-mesh 24x8192 @ 8192x512)
// grid = (C/16, 2, B), block = 32. M rows clamped (24 of 32 valid).
// ---------------------------------------------------------------------------
__global__ __launch_bounds__(32) void hm_x5_gemm(const float* __restrict__ hm,
                                                 const _Float16* __restrict__ x5,
                                                 float* __restrict__ x6,
                                                 int V, int K, int C) {
  int lane = threadIdx.x;
  int tileN = blockIdx.x * 16;
  int tileM = blockIdx.y * 16;
  int b = blockIdx.z;
  int l16 = lane & 15, half = lane >> 4;
  int m = tileM + l16;
  int mc = (m < K) ? m : 0;  // clamp
  int n = tileN + l16;
  const float* hb = hm + (long)b * V * K;
  const _Float16* xb = x5 + (long)b * V * C;
  v8f acc = {};
  for (int k0 = 0; k0 < V; k0 += 32) {
    v16h a, bf;
#pragma unroll
    for (int vg = 0; vg < 8; ++vg) {
      int kb = (vg < 4) ? 0 : 16;
      int vv = vg & 3;
#pragma unroll
      for (int e2 = 0; e2 < 2; ++e2) {
        int k = k0 + kb + half * 8 + vv * 2 + e2;
        a[2 * vg + e2]  = (_Float16)hb[(long)k * K + mc];
        bf[2 * vg + e2] = xb[(long)k * C + n];
      }
    }
    acc = __builtin_amdgcn_wmma_f32_16x16x32_f16(false, a, false, bf, (short)0, acc,
                                                 false, false);
  }
#pragma unroll
  for (int r = 0; r < 8; ++r) {
    int mr = tileM + r + half * 8;
    if (mr < K) x6[((long)b * K + mr) * C + tileN + l16] = acc[r];
  }
}

// ---------------------------------------------------------------------------
// host orchestration
// ---------------------------------------------------------------------------
extern "C" void kernel_launch(void* const* d_in, const int* in_sizes, int n_in,
                              void* d_out, int out_size, void* d_ws, size_t ws_size,
                              hipStream_t stream) {
  const float* pos   = (const float*)d_in[0];
  const float* hm    = (const float*)d_in[1];
  const int*   eidx  = (const int*)d_in[2];
  const int*   batch = (const int*)d_in[3];
  const long N = in_sizes[3];
  const long E = in_sizes[2] / 2;
  const int  B = 8, K = 24;
  const int  V = (int)(N / B);
  const int* srcI = eidx;
  const int* dstI = eidx + E;
  auto P = [&](int i) { return (const float*)d_in[i]; };

  // workspace carve
  char* w = (char*)d_ws;
  size_t off = 0;
  auto carve = [&](size_t bytes) {
    char* p = w + off;
    off += (bytes + 255) & ~(size_t)255;
    return (void*)p;
  };
  _Float16* e_z1  = (_Float16*)carve((size_t)E * 256 * 2);
  _Float16* e_z2  = (_Float16*)carve((size_t)E * 256 * 2);
  unsigned* enc   = (unsigned*)carve((size_t)N * 256 * 4);
  _Float16* ntmp  = (_Float16*)carve((size_t)N * 256 * 2);
  _Float16* feat0 = (_Float16*)carve((size_t)E * 64 * 2);
  _Float16* x1    = (_Float16*)carve((size_t)N * 64 * 2);
  _Float16* x2    = (_Float16*)carve((size_t)N * 128 * 2);
  _Float16* x3    = (_Float16*)carve((size_t)N * 256 * 2);
  _Float16* x123  = (_Float16*)carve((size_t)N * 448 * 2);
  _Float16* x4    = (_Float16*)carve((size_t)N * 256 * 2);
  _Float16* x5    = (_Float16*)carve((size_t)N * 512 * 2);
  _Float16* xgin  = (_Float16*)carve((size_t)B * 448 * 2);
  _Float16* xg    = (_Float16*)carve((size_t)B * 256 * 2);
  float* x6    = (float*)carve((size_t)B * K * 512 * 4);
  float* z6    = (float*)carve((size_t)B * K * 256 * 4);
  float* stats = (float*)carve(512 * 4);
  float* ss    = (float*)carve(512 * 4);
  // f16 weights
  _Float16* W0p  = (_Float16*)carve(64 * 64 * 2);
  _Float16* W1h  = (_Float16*)carve(64 * 64 * 2);
  _Float16* Wm1h = (_Float16*)carve(64 * 64 * 2);
  _Float16* W2ah = (_Float16*)carve(128 * 128 * 2);
  _Float16* W2bh = (_Float16*)carve(128 * 128 * 2);
  _Float16* Wm2h = (_Float16*)carve(128 * 128 * 2);
  _Float16* W3ah = (_Float16*)carve(256 * 256 * 2);
  _Float16* W3bh = (_Float16*)carve(256 * 256 * 2);
  _Float16* Wm3h = (_Float16*)carve(256 * 256 * 2);
  _Float16* Wp1h = (_Float16*)carve(256 * 448 * 2);
  _Float16* Wgh  = (_Float16*)carve(256 * 448 * 2);

  auto zero32 = [&](void* p, long n) {
    zero32_kernel<<<dim3((unsigned)((n + 255) / 256)), 256, 0, stream>>>((unsigned*)p, n);
  };
  auto cvt = [&](int pw, _Float16* dst, long rows, int Kin, int KinPad) {
    long n = rows * KinPad;
    cvt_w_kernel<<<dim3((unsigned)((n + 255) / 256)), 256, 0, stream>>>(P(pw), dst, n, Kin,
                                                                        KinPad);
  };
  auto finalize = [&](int pw, float invM, int C) {
    bn_finalize_kernel<<<dim3((unsigned)((C + 63) / 64)), 64, 0, stream>>>(
        stats, P(pw + 2), P(pw + 3), ss, invM, C);
  };
  auto gemm = [&](const _Float16* A, const _Float16* Wh, int pw, _Float16* Z, long M, int Kin,
                  int Cout) {
    zero32(stats, 2 * Cout);
    gemm_bias_relu_stats<<<dim3((unsigned)((M + 127) / 128), (unsigned)(Cout / 64)), 256, 0,
                           stream>>>(A, Wh, P(pw + 1), Z, stats, M, Kin, Cout);
    finalize(pw, 1.f / (float)M, Cout);
  };
  auto egemm = [&](const _Float16* X, const _Float16* Wh, int pw, _Float16* Z, int Cin,
                   int Cout) {
    zero32(stats, 2 * Cout);
    edge_gemm_bias_relu_stats<<<dim3((unsigned)((E + 127) / 128), (unsigned)(Cout / 64)), 256,
                                0, stream>>>(X, srcI, dstI, Wh, P(pw + 1), Z, stats, E, Cin,
                                             Cout);
    finalize(pw, 1.f / (float)E, Cout);
  };
  auto bn_apply = [&](_Float16* z, long n, int C) {
    bn_apply_kernel<<<dim3((unsigned)((n / 8 + 255) / 256)), 256, 0, stream>>>(z, ss, n / 8, C);
  };
  auto scatter = [&](const _Float16* z, int c) {
    zero32(enc, N * c);
    bn_scatter_max_kernel<<<dim3((unsigned)((E * c + 255) / 256)), 256, 0, stream>>>(
        z, ss, dstI, enc, E, c);
    decode_max_kernel<<<dim3((unsigned)((N * c + 255) / 256)), 256, 0, stream>>>(enc, ntmp,
                                                                                 N * c);
  };

  // ---- weight conversion (once per launch) ----
  cvt(4, W0p, 64, 6, 64);
  cvt(8, W1h, 64, 64, 64);
  cvt(12, Wm1h, 64, 64, 64);
  cvt(16, W2ah, 128, 128, 128);
  cvt(20, W2bh, 128, 128, 128);
  cvt(24, Wm2h, 128, 128, 128);
  cvt(28, W3ah, 256, 256, 256);
  cvt(32, W3bh, 256, 256, 256);
  cvt(36, Wm3h, 256, 256, 256);
  cvt(40, Wp1h, 256, 448, 448);
  cvt(44, Wgh, 256, 448, 448);

  // ---- gcu1 (cin 3 -> 64) ----
  build_feat0_kernel<<<dim3((unsigned)((E * 64 + 255) / 256)), 256, 0, stream>>>(pos, srcI,
                                                                                 dstI, feat0, E);
  gemm(feat0, W0p, 4, e_z1, E, 64, 64);
  bn_apply(e_z1, E * 64, 64);
  gemm(e_z1, W1h, 8, e_z2, E, 64, 64);
  scatter(e_z2, 64);
  gemm(ntmp, Wm1h, 12, x1, N, 64, 64);
  bn_apply(x1, N * 64, 64);

  // ---- gcu2 (64 -> 128) ----
  egemm(x1, W2ah, 16, e_z1, 64, 128);
  bn_apply(e_z1, E * 128, 128);
  gemm(e_z1, W2bh, 20, e_z2, E, 128, 128);
  scatter(e_z2, 128);
  gemm(ntmp, Wm2h, 24, x2, N, 128, 128);
  bn_apply(x2, N * 128, 128);

  // ---- gcu3 (128 -> 256) ----
  egemm(x2, W3ah, 28, e_z1, 128, 256);
  bn_apply(e_z1, E * 256, 256);
  gemm(e_z1, W3bh, 32, e_z2, E, 256, 256);
  scatter(e_z2, 256);
  gemm(ntmp, Wm3h, 36, x3, N, 256, 256);
  bn_apply(x3, N * 256, 256);

  // ---- concat, mlp1, global max + mlp_glb, x5 ----
  concat3_kernel<<<dim3((unsigned)((N * 56 + 255) / 256)), 256, 0, stream>>>(x1, x2, x3, x123,
                                                                             N);
  gemm(x123, Wp1h, 40, x4, N, 448, 256);
  bn_apply(x4, N * 256, 256);

  mesh_maxpool_kernel<<<dim3(448, B), 256, 0, stream>>>(x123, xgin, V, 448);
  gemm(xgin, Wgh, 44, xg, B, 448, 256);
  bn_apply(xg, (long)B * 256, 256);

  build_x5_kernel<<<dim3((unsigned)((N * 64 + 255) / 256)), 256, 0, stream>>>(x4, xg, batch,
                                                                              x5, N);

  // ---- hm contraction (WMMA), final linear + relu + BN -> d_out ----
  hm_x5_gemm<<<dim3(512 / 16, 2, B), 32, 0, stream>>>(hm, x5, x6, V, K, 512);

  zero32(stats, 2 * 256);
  final_linear_kernel<<<dim3((unsigned)((B * K * 256 + 255) / 256)), 256, 0, stream>>>(
      x6, P(48), P(49), z6, stats, B * K, 512, 256);
  bn_finalize_kernel<<<dim3(4), 64, 0, stream>>>(stats, P(50), P(51), ss,
                                                 1.f / (float)(B * K), 256);
  bn_out_kernel<<<dim3((unsigned)((B * K * 256 + 255) / 256)), 256, 0, stream>>>(
      z6, ss, (float*)d_out, B * K * 256, 256);
}
